// Model_UMI_27333171872582
// MI455X (gfx1250) — matrix operand order repaired and checked by reading, not compiled, not run on hardware
//
#include <hip/hip_runtime.h>

typedef __bf16        v16bf __attribute__((ext_vector_type(16)));
typedef float         v8f   __attribute__((ext_vector_type(8)));
typedef float         v4f   __attribute__((ext_vector_type(4)));
typedef unsigned int  v4u   __attribute__((ext_vector_type(4)));
typedef unsigned int  v2u   __attribute__((ext_vector_type(2)));

#define B_   1024
#define S_   256
#define D_   192
#define H4_  768
#define K_   4
#define XS_STRIDE 200   // ushorts per row: 192 data + 8 pad (100 dwords -> conflict-free column reads)

__device__ __forceinline__ unsigned short f2bf(float f) {
  unsigned int u = __float_as_uint(f);
  u += 0x7FFFu + ((u >> 16) & 1u);     // round-to-nearest-even
  return (unsigned short)(u >> 16);
}

__device__ __forceinline__ float fast_tanh(float x) {
  x = fminf(fmaxf(x, -15.f), 15.f);
  float e = __expf(2.0f * x);
  return __fdividef(e - 1.0f, e + 1.0f);   // v_rcp_f32 + mul, no IEEE div expansion
}

union ABu { v16bf v; v4u q[2]; };

__global__ void prep_w1t(const float* __restrict__ W1, unsigned short* __restrict__ w1t) {
  int i = blockIdx.x * 256 + threadIdx.x;
  if (i < H4_ * D_) {
    int n = i / D_, k = i % D_;
    w1t[i] = f2bf(W1[k * H4_ + n]);   // W1t[n][k] = W1[k][n], bf16
  }
}

__global__ __launch_bounds__(256) void umi_kernel(
    const float* __restrict__ his, const int* __restrict__ mask,
    const float* __restrict__ uid, const float* __restrict__ item,
    const float* __restrict__ pos, const unsigned short* __restrict__ w1t,
    const float* __restrict__ b1,  const float* __restrict__ W2,
    const float* __restrict__ b2,  const float* __restrict__ uaW1,
    const float* __restrict__ uab1,const float* __restrict__ uaW2,
    const float* __restrict__ uab2,const float* __restrict__ fc1W,
    const float* __restrict__ fc1b,const float* __restrict__ fc2W,
    const float* __restrict__ fc2b,float* __restrict__ out)
{
  __shared__ unsigned short xs[S_ * XS_STRIDE];   // 102400 B, x = his+pos in bf16
  __shared__ float wls[K_ * S_];                  // w, then softmax a
  __shared__ float inter[K_ * D_];
  __shared__ float g1ls[32];
  __shared__ float g2ls[12];
  __shared__ float concls[K_ * 2 * D_];
  __shared__ float u1ls[K_ * 2 * D_];
  __shared__ float uuls[K_ * D_];
  __shared__ float lgls[K_];
  __shared__ int   sidx;

  const int t    = threadIdx.x;
  const int b    = blockIdx.x;
  const int lane = t & 31;
  const int wv   = t >> 5;
  const size_t hisBase = (size_t)b * (S_ * D_);

  // ---- Stage A: x = his[b] + pos -> bf16 in LDS -------------------------
  {
    const float4* h4 = (const float4*)(his + hisBase);
    const float4* p4 = (const float4*)pos;
    for (int i = t; i < (S_ * D_) / 4; i += 256) {
      float4 hv = h4[i], pv = p4[i];
      unsigned int lo = (unsigned int)f2bf(hv.x + pv.x) | ((unsigned int)f2bf(hv.y + pv.y) << 16);
      unsigned int hi = (unsigned int)f2bf(hv.z + pv.z) | ((unsigned int)f2bf(hv.w + pv.w) << 16);
      int e = i * 4, row = e / D_, col = e % D_;
      v2u pk = { lo, hi };
      *(v2u*)(&xs[row * XS_STRIDE + col]) = pk;
    }
  }
  __syncthreads();

  // ---- Stage B: WMMA GEMM [256x192]x[192x768] fused tanh(+b1) and @W2 ---
  {
    const int lrow = lane & 15;
    const int hh   = lane >> 4;
    const int mt0  = 2 * wv, mt1 = 2 * wv + 1;
    const unsigned short* x0 = &xs[(mt0 * 16 + lrow) * XS_STRIDE];
    const unsigned short* x1 = &xs[(mt1 * 16 + lrow) * XS_STRIDE];
    float wacc0[32], wacc1[32];
    #pragma unroll
    for (int i = 0; i < 32; ++i) { wacc0[i] = 0.f; wacc1[i] = 0.f; }

    // B double-buffer: prefetch tile kt+1 (or first tile of next row) while
    // WMMA consumes tile kt. 6 tiles/row (even) -> parity identical each row.
    ABu bbuf[2];
    {
      const unsigned short* bp0 = w1t + (size_t)lrow * D_ + hh * 16;
      bbuf[0].q[0] = *(const v4u*)(bp0);
      bbuf[0].q[1] = *(const v4u*)(bp0 + 8);
    }

    for (int nt = 0; nt < 48; ++nt) {
      v8f c0 = {0.f,0.f,0.f,0.f,0.f,0.f,0.f,0.f};
      v8f c1 = {0.f,0.f,0.f,0.f,0.f,0.f,0.f,0.f};
      const unsigned short* bpc = w1t + (size_t)(nt * 16 + lrow) * D_ + hh * 16;
      const unsigned short* bpn = bpc + 16 * D_;      // next row's kt=0 tile

      // issue epilogue operand loads early; they retire under the WMMAs
      const int j = nt * 16 + lrow;
      const float b1v = b1[j];
      const v4f w2v = *(const v4f*)(W2 + 4 * j);

      #pragma unroll
      for (int kt = 0; kt < 6; ++kt) {
        const int cb = kt & 1, nb = cb ^ 1;
        if (kt < 5) {
          bbuf[nb].q[0] = *(const v4u*)(bpc + (kt + 1) * 32);
          bbuf[nb].q[1] = *(const v4u*)(bpc + (kt + 1) * 32 + 8);
        } else if (nt < 47) {
          bbuf[nb].q[0] = *(const v4u*)(bpn);
          bbuf[nb].q[1] = *(const v4u*)(bpn + 8);
        }
        const int kc = kt * 32 + hh * 16;    // A-tile K base for this half-wave
        ABu a0, a1;
        a0.q[0] = *(const v4u*)(x0 + kc);
        a0.q[1] = *(const v4u*)(x0 + kc + 8);
        a1.q[0] = *(const v4u*)(x1 + kc);
        a1.q[1] = *(const v4u*)(x1 + kc + 8);
        c0 = __builtin_amdgcn_wmma_f32_16x16x32_bf16(false, a0.v, false, bbuf[cb].v, (short)0, c0, false, false);
        c1 = __builtin_amdgcn_wmma_f32_16x16x32_bf16(false, a1.v, false, bbuf[cb].v, (short)0, c1, false, false);
      }

      // epilogue: lane holds column j = nt*16 + lrow for 8 rows per tile
      #pragma unroll
      for (int r = 0; r < 8; ++r) {
        float t0 = fast_tanh(c0[r] + b1v);
        float t1 = fast_tanh(c1[r] + b1v);
        #pragma unroll
        for (int k = 0; k < 4; ++k) {
          wacc0[r * 4 + k] += t0 * w2v[k];
          wacc1[r * 4 + k] += t1 * w2v[k];
        }
      }
    }
    // reduce the 16 j-lanes within each half-wave
    #pragma unroll
    for (int i = 0; i < 32; ++i) {
      #pragma unroll
      for (int m = 1; m < 16; m <<= 1) {
        wacc0[i] += __shfl_xor(wacc0[i], m, 32);
        wacc1[i] += __shfl_xor(wacc1[i], m, 32);
      }
    }
    if (lrow < 4) {
      const int k = lrow;
      #pragma unroll
      for (int r = 0; r < 8; ++r) {
        wls[k * S_ + mt0 * 16 + hh * 8 + r] = wacc0[r * 4 + k] + b2[k];
        wls[k * S_ + mt1 * 16 + hh * 8 + r] = wacc1[r * 4 + k] + b2[k];
      }
    }
  }
  __syncthreads();

  // ---- masked softmax over S, per head k (waves 0..3) -------------------
  if (wv < 4) {
    const int k = wv;
    float v[8];
    float mx = -3.0e38f;
    #pragma unroll
    for (int i = 0; i < 8; ++i) {
      int s = lane + 32 * i;
      float val = wls[k * S_ + s];
      if (mask[b * S_ + s] == 0) val = -4294967295.0f;
      v[i] = val;
      mx = fmaxf(mx, val);
    }
    #pragma unroll
    for (int m = 1; m < 32; m <<= 1) mx = fmaxf(mx, __shfl_xor(mx, m, 32));
    float sum = 0.f;
    #pragma unroll
    for (int i = 0; i < 8; ++i) { v[i] = __expf(v[i] - mx); sum += v[i]; }
    #pragma unroll
    for (int m = 1; m < 32; m <<= 1) sum += __shfl_xor(sum, m, 32);
    const float inv = 1.0f / sum;
    #pragma unroll
    for (int i = 0; i < 8; ++i) wls[k * S_ + lane + 32 * i] = v[i] * inv;
  }
  __syncthreads();

  // ---- interest = a @ his[b]  (L2-hot f32) ------------------------------
  {
    const float* hb = his + hisBase;
    for (int oi = t; oi < K_ * D_; oi += 256) {
      const int k = oi / D_, d = oi % D_;
      float acc = 0.f;
      for (int s = 0; s < S_; ++s) acc += wls[k * S_ + s] * hb[s * D_ + d];
      inter[oi] = acc;
    }
  }
  __syncthreads();

  // ---- gating MLP -------------------------------------------------------
  const float* uidb = uid + b * D_;
  if (t < 32) {
    const int k = t >> 3, o = t & 7;
    float acc = uab1[o];
    for (int j = 0; j < D_; ++j) acc += uidb[j] * uaW1[j * 8 + o];
    for (int j = 0; j < D_; ++j) acc += inter[k * D_ + j] * uaW1[(D_ + j) * 8 + o];
    g1ls[t] = fmaxf(acc, 0.f);
  }
  __syncthreads();
  if (t < 12) {
    const int k = t / 3, o = t % 3;
    float acc = uab2[o];
    #pragma unroll
    for (int j = 0; j < 8; ++j) acc += g1ls[k * 8 + j] * uaW2[j * 3 + o];
    g2ls[t] = 1.0f / (1.0f + __expf(-acc));
  }
  __syncthreads();
  for (int ci = t; ci < K_ * 2 * D_; ci += 256) {
    const int k = ci / (2 * D_), c = ci % (2 * D_);
    concls[ci] = (c < D_) ? uidb[c] * g2ls[k * 3 + (c / 64)]
                          : inter[k * D_ + (c - D_)];
  }
  __syncthreads();

  // ---- fc1: u1 = conc @ fc1W + fc1b (weight loaded once per 4 heads) ----
  for (int c = t; c < 2 * D_; c += 256) {
    float a0 = fc1b[c], a1 = a0, a2 = a0, a3 = a0;
    for (int j = 0; j < 2 * D_; ++j) {
      const float w = fc1W[j * (2 * D_) + c];
      a0 += concls[0 * 384 + j] * w;
      a1 += concls[1 * 384 + j] * w;
      a2 += concls[2 * 384 + j] * w;
      a3 += concls[3 * 384 + j] * w;
    }
    u1ls[0 * 384 + c] = a0; u1ls[1 * 384 + c] = a1;
    u1ls[2 * 384 + c] = a2; u1ls[3 * 384 + c] = a3;
  }
  __syncthreads();

  // ---- fc2: u = relu(u1 @ fc2W + fc2b) ----------------------------------
  for (int c = t; c < D_; c += 256) {
    float a0 = fc2b[c], a1 = a0, a2 = a0, a3 = a0;
    for (int j = 0; j < 2 * D_; ++j) {
      const float w = fc2W[j * D_ + c];
      a0 += u1ls[0 * 384 + j] * w;
      a1 += u1ls[1 * 384 + j] * w;
      a2 += u1ls[2 * 384 + j] * w;
      a3 += u1ls[3 * 384 + j] * w;
    }
    uuls[0 * D_ + c] = fmaxf(a0, 0.f);
    uuls[1 * D_ + c] = fmaxf(a1, 0.f);
    uuls[2 * D_ + c] = fmaxf(a2, 0.f);
    uuls[3 * D_ + c] = fmaxf(a3, 0.f);
  }
  __syncthreads();

  // ---- argmax routing (softmax is monotone -> argmax of logits) ---------
  if (t < 4) {
    const float* ib = item + b * D_;
    float acc = 0.f;
    for (int d = 0; d < D_; ++d) acc += uuls[t * D_ + d] * ib[d];
    lgls[t] = acc;
  }
  __syncthreads();
  if (t == 0) {
    int best = 0; float bv = lgls[0];
    #pragma unroll
    for (int k = 1; k < 4; ++k) if (lgls[k] > bv) { bv = lgls[k]; best = k; }
    sidx = best;
  }
  __syncthreads();
  if (t < D_) out[(size_t)b * D_ + t] = uuls[sidx * D_ + t];
}

extern "C" void kernel_launch(void* const* d_in, const int* in_sizes, int n_in,
                              void* d_out, int out_size, void* d_ws, size_t ws_size,
                              hipStream_t stream) {
  const float* his  = (const float*)d_in[0];
  const int*   mask = (const int*)d_in[1];
  const float* uid  = (const float*)d_in[2];
  const float* item = (const float*)d_in[3];
  const float* pos  = (const float*)d_in[4];
  const float* W1   = (const float*)d_in[5];
  const float* b1   = (const float*)d_in[6];
  const float* W2   = (const float*)d_in[7];
  const float* b2   = (const float*)d_in[8];
  const float* uaW1 = (const float*)d_in[9];
  const float* uab1 = (const float*)d_in[10];
  const float* uaW2 = (const float*)d_in[11];
  const float* uab2 = (const float*)d_in[12];
  const float* fc1W = (const float*)d_in[13];
  const float* fc1b = (const float*)d_in[14];
  const float* fc2W = (const float*)d_in[15];
  const float* fc2b = (const float*)d_in[16];
  float* out = (float*)d_out;

  unsigned short* w1t = (unsigned short*)d_ws;   // 768*192 bf16 = 294912 B

  prep_w1t<<<(H4_ * D_ + 255) / 256, 256, 0, stream>>>(W1, w1t);
  umi_kernel<<<B_, 256, 0, stream>>>(his, mask, uid, item, pos, w1t, b1, W2, b2,
                                     uaW1, uab1, uaW2, uab2, fc1W, fc1b, fc2W, fc2b, out);
}
